// Model_20246475833862
// MI455X (gfx1250) — matrix operand order repaired and checked by reading, not compiled, not run on hardware
//
#include <hip/hip_runtime.h>
#include <math.h>

typedef __attribute__((ext_vector_type(16))) __bf16 v16bf;
typedef __attribute__((ext_vector_type(8)))  float  v8f;

#define FDIM 128

__device__ __forceinline__ float lrelu(float v) { return v >= 0.f ? v : 0.2f * v; }

// monotonic float <-> uint mapping for atomicMax on floats
__device__ __forceinline__ unsigned fenc(float f) {
    unsigned u = __float_as_uint(f);
    return (u & 0x80000000u) ? ~u : (u | 0x80000000u);
}
__device__ __forceinline__ float fdec(unsigned u) {
    return __uint_as_float((u & 0x80000000u) ? (u & 0x7fffffffu) : ~u);
}

// ---------------- weight pre-pack into WMMA B-fragment layout ----------------
// W: [128, ncols] row-major fp32  ->  P[(tile*4 + ks)*32 + lane] = v16bf fragment
// lane half 0 holds K = ks*32 + 0..15, half 1 holds K = ks*32 + 16..31, col = tile*16 + lane%16
__global__ void pack_b_frags(const float* __restrict__ W, v16bf* __restrict__ P, int ncols) {
    int wid = blockIdx.x * (blockDim.x >> 5) + (threadIdx.x >> 5);
    int ntiles = ncols >> 4;
    if (wid >= ntiles * 4) return;
    int tile = wid >> 2, ks = wid & 3;
    int lane = threadIdx.x & 31;
    int half = lane >> 4;
    const float* bp = W + (size_t)(ks * 32 + half * 16) * ncols + tile * 16 + (lane & 15);
    v16bf r;
#pragma unroll
    for (int i = 0; i < 16; ++i) r[i] = (__bf16)bp[(size_t)i * ncols];
    P[(size_t)wid * 32 + lane] = r;
}

// ---------------- WMMA GEMM (fp32 in/out, bf16 multiply, K=128) ----------------
// A: [rows,128] row-major fp32; Bp: pre-packed fragments; C: [rows,ncols] fp32
// one wave computes a 16-row strip; A fragments hoisted; two column tiles per
// iteration give two independent WMMA accumulation chains for pipelining.

__device__ __forceinline__ v16bf load_a_frag(const float* __restrict__ A,
                                             int row0, int kbase) {
    int lane = threadIdx.x & 31;
    int half = lane >> 4;
    const float* ap = A + (size_t)(row0 + (lane & 15)) * FDIM + kbase + half * 16;
    float4 a0 = *(const float4*)(ap);        // K offsets 0..3
    float4 a1 = *(const float4*)(ap + 4);    // 4..7
    float4 a2 = *(const float4*)(ap + 8);    // 8..11
    float4 a3 = *(const float4*)(ap + 12);   // 12..15
    v16bf r;
    r[0] = (__bf16)a0.x; r[1] = (__bf16)a0.y; r[2] = (__bf16)a0.z; r[3] = (__bf16)a0.w;
    r[4] = (__bf16)a1.x; r[5] = (__bf16)a1.y; r[6] = (__bf16)a1.z; r[7] = (__bf16)a1.w;
    r[8] = (__bf16)a2.x; r[9] = (__bf16)a2.y; r[10] = (__bf16)a2.z; r[11] = (__bf16)a2.w;
    r[12] = (__bf16)a3.x; r[13] = (__bf16)a3.y; r[14] = (__bf16)a3.z; r[15] = (__bf16)a3.w;
    return r;
}

__device__ __forceinline__ void store_c_tile(float* __restrict__ C, const v8f& acc,
                                             const float* __restrict__ bias,
                                             int row0, int ncols, int tile, int act) {
    int lane = threadIdx.x & 31;
    int half = lane >> 4;
    int col = tile * 16 + (lane & 15);
    float bv = bias ? bias[col] : 0.f;
#pragma unroll
    for (int r = 0; r < 8; ++r) {
        float v = acc[r] + bv;
        if (act) v = lrelu(v);
        C[(size_t)(row0 + half * 8 + r) * ncols + col] = v;
    }
}

__global__ void gemm_k128_wmma(const float* __restrict__ A, const v16bf* __restrict__ Bp,
                               const float* __restrict__ bias, float* __restrict__ C,
                               int rows, int ncols, int act) {
    int wave = blockIdx.x * (blockDim.x >> 5) + (threadIdx.x >> 5);
    int row0 = wave * 16;
    if (row0 >= rows) return;

    int lane = threadIdx.x & 31;

    v16bf afrag[4];
#pragma unroll
    for (int ks = 0; ks < 4; ++ks) afrag[ks] = load_a_frag(A, row0, ks * 32);

    int ntiles = ncols >> 4;           // always even (8 or 16)
    for (int tile = 0; tile < ntiles; tile += 2) {
        const v16bf* bp0 = Bp + ((size_t)tile * 4) * 32 + lane;
        const v16bf* bp1 = bp0 + 4 * 32;
        v16bf b0[4], b1[4];
#pragma unroll
        for (int ks = 0; ks < 4; ++ks) { b0[ks] = bp0[ks * 32]; b1[ks] = bp1[ks * 32]; }
        v8f acc0 = {}, acc1 = {};
#pragma unroll
        for (int ks = 0; ks < 4; ++ks) {
            acc0 = __builtin_amdgcn_wmma_f32_16x16x32_bf16(
                false, afrag[ks], false, b0[ks], (short)0, acc0, false, false);
            acc1 = __builtin_amdgcn_wmma_f32_16x16x32_bf16(
                false, afrag[ks], false, b1[ks], (short)0, acc1, false, false);
        }
        store_c_tile(C, acc0, bias, row0, ncols, tile, act);
        store_c_tile(C, acc1, bias, row0, ncols, tile + 1, act);
    }
}

// ---------------- BatchNorm (training-mode batch stats) ----------------
#define RPB 512
__global__ void colstats_kernel(const float* __restrict__ X, int rows,
                                float* __restrict__ sums) {
    __shared__ float sh[512];
    int col = threadIdx.x & 127;
    int sub = threadIdx.x >> 7;
    int row0 = blockIdx.x * RPB;
    int rend = min(rows, row0 + RPB);
    float s = 0.f, q = 0.f;
    for (int r = row0 + sub; r < rend; r += 2) {
        float v = X[(size_t)r * FDIM + col];
        s += v; q += v * v;
    }
    sh[threadIdx.x] = s; sh[256 + threadIdx.x] = q;
    __syncthreads();
    if (sub == 0) {
        atomicAdd(&sums[col], sh[col] + sh[col + 128]);
        atomicAdd(&sums[128 + col], sh[256 + col] + sh[256 + col + 128]);
    }
}

__global__ void bn_finalize(const float* __restrict__ sums, const float* __restrict__ g,
                            const float* __restrict__ b, int rows,
                            float* __restrict__ ss) {
    int c = threadIdx.x;
    float inv = 1.f / (float)rows;
    float mean = sums[c] * inv;
    float var = sums[128 + c] * inv - mean * mean;
    float sc = g[c] * rsqrtf(var + 1e-5f);
    ss[c] = sc;
    ss[128 + c] = b[c] - mean * sc;
}

__global__ void bn_apply(const float* __restrict__ X, const float* __restrict__ ss,
                         float* __restrict__ Y, size_t total) {
    size_t i = (size_t)blockIdx.x * blockDim.x + threadIdx.x;
    if (i >= total) return;
    int c = (int)(i & 127);
    Y[i] = X[i] * ss[c] + ss[128 + c];
}

// ---------------- segment / attention kernels ----------------

// ea[he_e[e], f] += X[he_n[e], f]
__global__ void scatter_edge_sum(const float* __restrict__ X, const int* __restrict__ he_n,
                                 const int* __restrict__ he_e, float* __restrict__ ea,
                                 size_t total) {
    size_t i = (size_t)blockIdx.x * blockDim.x + threadIdx.x;
    if (i >= total) return;
    int f = (int)(i & 127);
    size_t e = i >> 7;
    int n = he_n[e], m = he_e[e];
    atomicAdd(&ea[(size_t)m * FDIM + f], X[(size_t)n * FDIM + f]);
}

// s[row*heads+h] = dot(X[row, h*128 : h*128+128], att[h*256 + attOff + j])
__global__ void attn_dot(const float* __restrict__ X, const float* __restrict__ att,
                         float* __restrict__ s, int rows, int heads, int attOff) {
    int wid = blockIdx.x * (blockDim.x >> 5) + (threadIdx.x >> 5);
    int lane = threadIdx.x & 31;
    if (wid >= rows * heads) return;
    int row = wid / heads, h = wid % heads;
    const float* xp = X + ((size_t)row * heads + h) * FDIM;
    const float* ap = att + h * 256 + attOff;
    float p = 0.f;
#pragma unroll
    for (int j = 0; j < 4; ++j) p += xp[lane + 32 * j] * ap[lane + 32 * j];
#pragma unroll
    for (int off = 16; off; off >>= 1) p += __shfl_xor(p, off, 32);
    if (lane == 0) s[wid] = p;
}

__global__ void fill_u32(unsigned* __restrict__ p, unsigned v, size_t n) {
    size_t i = (size_t)blockIdx.x * blockDim.x + threadIdx.x;
    if (i < n) p[i] = v;
}

__global__ void attn_max(const float* __restrict__ s1, const float* __restrict__ s2,
                         const int* __restrict__ he_n, const int* __restrict__ he_e,
                         unsigned* __restrict__ amax, size_t total, int heads) {
    size_t i = (size_t)blockIdx.x * blockDim.x + threadIdx.x;
    if (i >= total) return;
    size_t e = i / heads; int h = (int)(i % heads);
    int n = he_n[e], m = he_e[e];
    float v = lrelu(s1[(size_t)n * heads + h] + s2[(size_t)m * heads + h]);
    atomicMax(&amax[(size_t)n * heads + h], fenc(v));
}

__global__ void attn_exp(const float* __restrict__ s1, const float* __restrict__ s2,
                         const int* __restrict__ he_n, const int* __restrict__ he_e,
                         const unsigned* __restrict__ amax, float* __restrict__ esum,
                         float* __restrict__ attn, const float* __restrict__ he_w,
                         float* __restrict__ Dn, float* __restrict__ Bn,
                         size_t total, int heads) {
    size_t i = (size_t)blockIdx.x * blockDim.x + threadIdx.x;
    if (i >= total) return;
    size_t e = i / heads; int h = (int)(i % heads);
    int n = he_n[e], m = he_e[e];
    float v = lrelu(s1[(size_t)n * heads + h] + s2[(size_t)m * heads + h]);
    float ex = expf(v - fdec(amax[(size_t)n * heads + h]));
    attn[i] = ex;
    atomicAdd(&esum[(size_t)n * heads + h], ex);
    if (h == 0) {
        atomicAdd(&Dn[n], he_w[m]);
        atomicAdd(&Bn[m], 1.0f);
    }
}

__global__ void attn_norm(float* __restrict__ attn, const float* __restrict__ esum,
                          const int* __restrict__ he_n, size_t total, int heads) {
    size_t i = (size_t)blockIdx.x * blockDim.x + threadIdx.x;
    if (i >= total) return;
    size_t e = i / heads; int h = (int)(i % heads);
    float s = esum[(size_t)he_n[e] * heads + h];
    attn[i] = attn[i] / s;
}

__global__ void recip_inplace(float* __restrict__ p, size_t n) {
    size_t i = (size_t)blockIdx.x * blockDim.x + threadIdx.x;
    if (i >= n) return;
    float v = p[i];
    p[i] = v > 0.f ? 1.f / v : 0.f;
}

// eo[m,h,f] += Binv[m] * attn[e,h] * xl[n,h,f]
__global__ void msg_node2edge(const float* __restrict__ xl, const float* __restrict__ attn,
                              const float* __restrict__ Binv, const int* __restrict__ he_n,
                              const int* __restrict__ he_e, float* __restrict__ eo,
                              size_t total, int heads) {
    size_t i = (size_t)blockIdx.x * blockDim.x + threadIdx.x;
    if (i >= total) return;
    int f = (int)(i & 127);
    size_t eh = i >> 7;
    size_t e = eh / heads; int h = (int)(eh % heads);
    int n = he_n[e], m = he_e[e];
    float v = Binv[m] * attn[eh] * xl[((size_t)n * heads + h) * FDIM + f];
    atomicAdd(&eo[((size_t)m * heads + h) * FDIM + f], v);
}

// out[n,h,f] += Dinv[n] * attn[e,h] * eo[m,h,f]
__global__ void msg_edge2node(const float* __restrict__ eo, const float* __restrict__ attn,
                              const float* __restrict__ Dinv, const int* __restrict__ he_n,
                              const int* __restrict__ he_e, float* __restrict__ out,
                              size_t total, int heads) {
    size_t i = (size_t)blockIdx.x * blockDim.x + threadIdx.x;
    if (i >= total) return;
    int f = (int)(i & 127);
    size_t eh = i >> 7;
    size_t e = eh / heads; int h = (int)(eh % heads);
    int n = he_n[e], m = he_e[e];
    float v = Dinv[n] * attn[eh] * eo[((size_t)m * heads + h) * FDIM + f];
    atomicAdd(&out[((size_t)n * heads + h) * FDIM + f], v);
}

// dst[n,f] = a[n,f] + mean_h(outv[n,h,f]) + bias[f]
__global__ void residual_combine(const float* __restrict__ a, const float* __restrict__ outv,
                                 const float* __restrict__ bias, float* __restrict__ dst,
                                 size_t total, int heads) {
    size_t i = (size_t)blockIdx.x * blockDim.x + threadIdx.x;
    if (i >= total) return;
    int f = (int)(i & 127);
    size_t n = i >> 7;
    float o = 0.f;
    for (int h = 0; h < heads; ++h) o += outv[((size_t)n * heads + h) * FDIM + f];
    o *= (1.0f / (float)heads);
    float bb = bias ? bias[f] : 0.f;
    dst[i] = a[i] + o + bb;
}

// ---------------- host-side orchestration ----------------

static inline unsigned cdiv(size_t a, size_t b) { return (unsigned)((a + b - 1) / b); }

extern "C" void kernel_launch(void* const* d_in, const int* in_sizes, int n_in,
                              void* d_out, int out_size, void* d_ws, size_t ws_size,
                              hipStream_t stream) {
    (void)n_in; (void)out_size; (void)ws_size;

    const float* x      = (const float*)d_in[0];
    const int*   he_n   = (const int*)  d_in[1];
    const int*   he_e   = (const int*)  d_in[2];
    const float* he_w   = (const float*)d_in[3];
    const float* lin1_w = (const float*)d_in[4];
    const float* lin1_b = (const float*)d_in[5];
    const float* bn1_g  = (const float*)d_in[6];
    const float* bn1_b  = (const float*)d_in[7];
    const float* h1_w   = (const float*)d_in[8];
    const float* h1_att = (const float*)d_in[9];
    const float* h1_b   = (const float*)d_in[10];
    const float* bn2_g  = (const float*)d_in[11];
    const float* bn2_b  = (const float*)d_in[12];
    const float* h2_w   = (const float*)d_in[13];
    const float* h2_att = (const float*)d_in[14];
    const float* h2_b   = (const float*)d_in[15];
    const float* bn3_g  = (const float*)d_in[16];
    const float* bn3_b  = (const float*)d_in[17];
    const float* lin2_w = (const float*)d_in[18];
    const float* lin2_b = (const float*)d_in[19];
    const float* bn4_g  = (const float*)d_in[20];
    const float* bn4_b  = (const float*)d_in[21];

    const int N = in_sizes[0] / FDIM;
    const int E = in_sizes[1];
    const int M = in_sizes[3];

    // workspace layout
    char* w = (char*)d_ws;
    size_t off = 0;
    auto alloc = [&](size_t bytes) -> void* {
        void* p = w + off;
        off = (off + bytes + 255) & ~(size_t)255;
        return p;
    };
    float*    B0    = (float*)   alloc((size_t)N * FDIM * 4);   // current features
    float*    B1    = (float*)   alloc((size_t)N * FDIM * 4);   // pre-BN / out accum
    float*    B2    = (float*)   alloc((size_t)N * 256 * 4);    // xl / out accum (2-head)
    float*    EA    = (float*)   alloc((size_t)M * FDIM * 4);   // hyperedge attr
    float*    EAL   = (float*)   alloc((size_t)M * 256 * 4);    // lin(hyperedge attr)
    float*    EO    = (float*)   alloc((size_t)M * 256 * 4);    // edge aggregation
    float*    S1    = (float*)   alloc((size_t)N * 2 * 4);
    float*    S2    = (float*)   alloc((size_t)M * 2 * 4);
    unsigned* AMAX  = (unsigned*)alloc((size_t)N * 2 * 4);
    float*    ESUM  = (float*)   alloc((size_t)N * 2 * 4);
    float*    ATTN  = (float*)   alloc((size_t)E * 2 * 4);
    float*    DN    = (float*)   alloc((size_t)N * 4);
    float*    BN_   = (float*)   alloc((size_t)M * 4);
    float*    STATS = (float*)   alloc(256 * 4);
    float*    SS    = (float*)   alloc(256 * 4);
    v16bf*    PB256 = (v16bf*)   alloc((size_t)128 * 256 * 2);  // packed 128x256 weights
    v16bf*    PB128 = (v16bf*)   alloc((size_t)128 * 128 * 2);  // packed 128x128 weights

    const size_t NF = (size_t)N * FDIM;
    const unsigned TPB = 256;

    auto run_pack = [&](const float* W, v16bf* P, int ncols) {
        unsigned frags = (unsigned)(ncols >> 4) * 4;   // (tiles)*(ksteps) waves
        pack_b_frags<<<cdiv(frags, 8), TPB, 0, stream>>>(W, P, ncols);
    };

    auto run_bn = [&](const float* src, const float* g, const float* b, float* dst) {
        hipMemsetAsync(STATS, 0, 256 * 4, stream);
        colstats_kernel<<<cdiv(N, RPB), TPB, 0, stream>>>(src, N, STATS);
        bn_finalize<<<1, 128, 0, stream>>>(STATS, g, b, N, SS);
        bn_apply<<<cdiv(NF, TPB), TPB, 0, stream>>>(src, SS, dst, NF);
    };

    auto run_gemm = [&](const float* A, const v16bf* Bp, const float* bias, float* C,
                        int rows, int ncols, int act) {
        unsigned strips = (unsigned)(rows + 15) / 16;
        gemm_k128_wmma<<<cdiv(strips, 8), TPB, 0, stream>>>(A, Bp, bias, C, rows, ncols, act);
    };

    auto run_conv = [&](const float* xin, const v16bf* Wp, const float* att,
                        const float* bias, int heads, float* xl, float* eal,
                        float* out_accum, float* dst) {
        const size_t EH  = (size_t)E * heads;
        const size_t EHF = EH * FDIM;
        // xl = x @ W
        run_gemm(xin, Wp, nullptr, xl, N, heads * FDIM, 0);
        // hyperedge attr = segment_sum(x[he_n], he_e)
        hipMemsetAsync(EA, 0, (size_t)M * FDIM * 4, stream);
        scatter_edge_sum<<<cdiv((size_t)E * FDIM, TPB), TPB, 0, stream>>>(
            xin, he_n, he_e, EA, (size_t)E * FDIM);
        // eal = ea @ W
        run_gemm(EA, Wp, nullptr, eal, M, heads * FDIM, 0);
        // scalar attention scores (linearity of the att dot-product)
        attn_dot<<<cdiv((size_t)N * heads, 8), TPB, 0, stream>>>(xl, att, S1, N, heads, 0);
        attn_dot<<<cdiv((size_t)M * heads, 8), TPB, 0, stream>>>(eal, att, S2, M, heads, FDIM);
        // segment softmax over he_n
        fill_u32<<<cdiv((size_t)N * heads, TPB), TPB, 0, stream>>>(
            AMAX, 0x007FFFFFu /* fenc(-inf) */, (size_t)N * heads);
        attn_max<<<cdiv(EH, TPB), TPB, 0, stream>>>(S1, S2, he_n, he_e, AMAX, EH, heads);
        hipMemsetAsync(ESUM, 0, (size_t)N * heads * 4, stream);
        hipMemsetAsync(DN, 0, (size_t)N * 4, stream);
        hipMemsetAsync(BN_, 0, (size_t)M * 4, stream);
        attn_exp<<<cdiv(EH, TPB), TPB, 0, stream>>>(S1, S2, he_n, he_e, AMAX, ESUM, ATTN,
                                                    he_w, DN, BN_, EH, heads);
        attn_norm<<<cdiv(EH, TPB), TPB, 0, stream>>>(ATTN, ESUM, he_n, EH, heads);
        recip_inplace<<<cdiv((size_t)N, TPB), TPB, 0, stream>>>(DN, (size_t)N);
        recip_inplace<<<cdiv((size_t)M, TPB), TPB, 0, stream>>>(BN_, (size_t)M);
        // node -> hyperedge
        hipMemsetAsync(EO, 0, (size_t)M * heads * FDIM * 4, stream);
        msg_node2edge<<<cdiv(EHF, TPB), TPB, 0, stream>>>(xl, ATTN, BN_, he_n, he_e, EO,
                                                          EHF, heads);
        // hyperedge -> node
        hipMemsetAsync(out_accum, 0, (size_t)N * heads * FDIM * 4, stream);
        msg_edge2node<<<cdiv(EHF, TPB), TPB, 0, stream>>>(EO, ATTN, DN, he_n, he_e,
                                                          out_accum, EHF, heads);
        // dst = xin + mean_h(out) + bias
        residual_combine<<<cdiv(NF, TPB), TPB, 0, stream>>>(xin, out_accum, bias, dst,
                                                            NF, heads);
    };

    // ---- stage 1: x = BN1(lrelu(x @ lin1_w + lin1_b)) ----
    run_pack(lin1_w, PB128, FDIM);
    run_gemm(x, PB128, lin1_b, B1, N, FDIM, 1);
    run_bn(B1, bn1_g, bn1_b, B0);

    // ---- stage 2: hgconv1 (2 heads) + residual, BN2 ----
    run_pack(h1_w, PB256, 256);
    run_conv(B0, PB256, h1_att, h1_b, 2, B2, EAL, B2, B1);
    run_bn(B1, bn2_g, bn2_b, B0);

    // ---- stage 3: hgconv2 (1 head) + residual, BN3 ----
    run_pack(h2_w, PB128, FDIM);
    run_conv(B0, PB128, h2_att, h2_b, 1, B2, EAL, B1, B1);
    run_bn(B1, bn3_g, bn3_b, B0);

    // ---- stage 4: x = lrelu(x @ lin2_w + lin2_b); BN4(src + x) -> out ----
    run_pack(lin2_w, PB128, FDIM);
    run_gemm(B0, PB128, lin2_b, B1, N, FDIM, 1);
    residual_combine<<<cdiv(NF, TPB), TPB, 0, stream>>>(x, B1, nullptr, B1, NF, 1);
    hipMemsetAsync(STATS, 0, 256 * 4, stream);
    colstats_kernel<<<cdiv(N, RPB), TPB, 0, stream>>>(B1, N, STATS);
    bn_finalize<<<1, 128, 0, stream>>>(STATS, bn4_g, bn4_b, N, SS);
    bn_apply<<<cdiv(NF, TPB), TPB, 0, stream>>>(B1, SS, (float*)d_out, NF);
}
